// MultiHeadAttention_59081570124523
// MI455X (gfx1250) — matrix operand order repaired and checked by reading, not compiled, not run on hardware
//
#include <hip/hip_runtime.h>

typedef float v8f  __attribute__((ext_vector_type(8)));
typedef __bf16 v16bf __attribute__((ext_vector_type(16)));

#define NEGV (-1000000000.0f)

// Fragment: 16 bf16 = 32 bytes = 8 VGPRs, viewed as ushorts / two uint4.
union Frag {
    v16bf v;
    unsigned short s[16];
    uint4 q[2];
};

__device__ __forceinline__ v8f wmma_bf16(const Frag& a, const Frag& b, v8f c) {
    return __builtin_amdgcn_wmma_f32_16x16x32_bf16(false, a.v, false, b.v,
                                                   (short)0, c, false, false);
}

// ---- branch-free f32 -> bf16 RNE + exact-residual split (bit ops only) ----
__device__ __forceinline__ unsigned short f2bf(float x) {
    unsigned u = __float_as_uint(x);
    return (unsigned short)((u + 0x7FFFu + ((u >> 16) & 1u)) >> 16);
}
__device__ __forceinline__ float bf2f(unsigned short h) {
    return __uint_as_float(((unsigned)h) << 16);
}
__device__ __forceinline__ void split_bf(float x, unsigned short& hi, unsigned short& lo) {
    unsigned short h = f2bf(x);
    hi = h;
    lo = f2bf(x - bf2f(h));
}

// ---------------------------------------------------------------------------
// bf16x3 GEMM: C[M,N] = A[M,K]*B[K,N] + bias[N], fp32 in/out.
// 256 thr (8 waves), block tile 128x64, wave 32x32 (2x2 accs), K-step 32.
// A,B split to hi/lo bf16 during LDS staging; B staged transposed [n][k].
// 3 bf16 WMMAs (hi*hi + hi*lo + lo*hi) per 16x16x32 step.
// ---------------------------------------------------------------------------
__global__ __launch_bounds__(256)
void gemm_bf16x3(const float* __restrict__ A, const float* __restrict__ B,
                 const float* __restrict__ bias, float* __restrict__ C,
                 int M, int N, int K)
{
    __shared__ __align__(16) unsigned short sAhi[128 * 32];
    __shared__ __align__(16) unsigned short sAlo[128 * 32];
    __shared__ __align__(16) unsigned short sBhi[64 * 32];   // [n][k]
    __shared__ __align__(16) unsigned short sBlo[64 * 32];

    const int tid  = threadIdx.x;
    const int wave = tid >> 5, lane = tid & 31;
    const int wm   = wave >> 1, wn = wave & 1;
    const int hf   = lane >> 4, l16 = lane & 15;
    const int m_blk = blockIdx.x * 128;
    const int n_blk = blockIdx.y * 64;

    v8f acc[2][2];
    for (int i = 0; i < 2; ++i)
        for (int j = 0; j < 2; ++j)
            acc[i][j] = v8f{};

    for (int k0 = 0; k0 < K; k0 += 32) {
        // ---- stage A 128x32: 1024 float4, 4/thread; split hi/lo ----
        #pragma unroll
        for (int i = 0; i < 4; ++i) {
            int t = tid + i * 256;
            int row = t >> 3, c = t & 7;
            float4 f = *reinterpret_cast<const float4*>(
                A + (size_t)(m_blk + row) * K + k0 + c * 4);
            unsigned short h0,h1,h2,h3,l0,l1,l2,l3;
            split_bf(f.x, h0, l0); split_bf(f.y, h1, l1);
            split_bf(f.z, h2, l2); split_bf(f.w, h3, l3);
            uint2 hv; hv.x = (unsigned)h0 | ((unsigned)h1 << 16);
                      hv.y = (unsigned)h2 | ((unsigned)h3 << 16);
            uint2 lv; lv.x = (unsigned)l0 | ((unsigned)l1 << 16);
                      lv.y = (unsigned)l2 | ((unsigned)l3 << 16);
            *reinterpret_cast<uint2*>(&sAhi[row * 32 + c * 4]) = hv;
            *reinterpret_cast<uint2*>(&sAlo[row * 32 + c * 4]) = lv;
        }
        // ---- stage B 32x64 transposed: 512 float4, 2/thread ----
        #pragma unroll
        for (int i = 0; i < 2; ++i) {
            int t = tid + i * 256;
            int kr = t >> 4, c = t & 15;
            float4 f = *reinterpret_cast<const float4*>(
                B + (size_t)(k0 + kr) * N + n_blk + c * 4);
            float fv[4] = {f.x, f.y, f.z, f.w};
            #pragma unroll
            for (int u = 0; u < 4; ++u) {
                int n = c * 4 + u;
                unsigned short h, l;
                split_bf(fv[u], h, l);
                sBhi[n * 32 + kr] = h;
                sBlo[n * 32 + kr] = l;
            }
        }
        // ---- prefetch next k-slab (global_prefetch_b8) ----
        if (k0 + 32 < K) {
            __builtin_prefetch(A + (size_t)(m_blk + (tid >> 3)) * K + k0 + 32 + (tid & 7) * 4, 0, 1);
            __builtin_prefetch(A + (size_t)(m_blk + 64 + (tid >> 3)) * K + k0 + 32 + (tid & 7) * 4, 0, 1);
            __builtin_prefetch(B + (size_t)(k0 + 32 + (tid >> 4)) * N + n_blk + (tid & 15) * 4, 0, 1);
            __builtin_prefetch(B + (size_t)(k0 + 48 + (tid >> 4)) * N + n_blk + (tid & 15) * 4, 0, 1);
        }
        __syncthreads();

        // ---- fragments (ISA A 16x32 / B 32x16 bf16 layouts) ----
        Frag aHi[2], aLo[2], bHi[2], bLo[2];
        #pragma unroll
        for (int mi = 0; mi < 2; ++mi) {
            int row = wm * 32 + mi * 16 + l16;
            aHi[mi].q[0] = *reinterpret_cast<uint4*>(&sAhi[row * 32 + hf * 8]);
            aHi[mi].q[1] = *reinterpret_cast<uint4*>(&sAhi[row * 32 + 16 + hf * 8]);
            aLo[mi].q[0] = *reinterpret_cast<uint4*>(&sAlo[row * 32 + hf * 8]);
            aLo[mi].q[1] = *reinterpret_cast<uint4*>(&sAlo[row * 32 + 16 + hf * 8]);
        }
        #pragma unroll
        for (int ni = 0; ni < 2; ++ni) {
            int col = wn * 32 + ni * 16 + l16;
            bHi[ni].q[0] = *reinterpret_cast<uint4*>(&sBhi[col * 32 + hf * 8]);
            bHi[ni].q[1] = *reinterpret_cast<uint4*>(&sBhi[col * 32 + 16 + hf * 8]);
            bLo[ni].q[0] = *reinterpret_cast<uint4*>(&sBlo[col * 32 + hf * 8]);
            bLo[ni].q[1] = *reinterpret_cast<uint4*>(&sBlo[col * 32 + 16 + hf * 8]);
        }
        #pragma unroll
        for (int mi = 0; mi < 2; ++mi)
            #pragma unroll
            for (int ni = 0; ni < 2; ++ni) {
                acc[mi][ni] = wmma_bf16(aHi[mi], bHi[ni], acc[mi][ni]);
                acc[mi][ni] = wmma_bf16(aHi[mi], bLo[ni], acc[mi][ni]);
                acc[mi][ni] = wmma_bf16(aLo[mi], bHi[ni], acc[mi][ni]);
            }
        __syncthreads();
    }

    for (int mi = 0; mi < 2; ++mi)
        for (int ni = 0; ni < 2; ++ni) {
            int col = n_blk + wn * 32 + ni * 16 + l16;
            float bv = bias[col];
            for (int r = 0; r < 8; ++r) {
                int row = m_blk + wm * 32 + mi * 16 + r + hf * 8;
                C[(size_t)row * N + col] = acc[mi][ni][r] + bv;
            }
        }
}

// ---------------------------------------------------------------------------
// V split+transpose pre-pass: qkv f32 [B,S,3D] -> vT hi/lo bf16 [B*H, hd, S].
// One block per (b,s) row; coalesced reads, scattered b16 writes (one-shot).
// ---------------------------------------------------------------------------
__global__ __launch_bounds__(256)
void split_v_transpose(const float* __restrict__ qkv,
                       unsigned short* __restrict__ vThi,
                       unsigned short* __restrict__ vTlo,
                       int S, int D, int H)
{
    const int bs = blockIdx.x;               // b*S + s
    const int s  = bs & (S - 1);
    const int b  = bs / S;
    const float* vrow = qkv + (size_t)bs * 3 * D + 2 * D;
    const int d0 = threadIdx.x * 4;
    float4 f = *reinterpret_cast<const float4*>(vrow + d0);
    float fv[4] = {f.x, f.y, f.z, f.w};
    #pragma unroll
    for (int u = 0; u < 4; ++u) {
        int d = d0 + u;
        int h = d >> 6, n = d & 63;
        size_t off = ((size_t)(b * H + h) * 64 + n) * S + s;
        unsigned short hi, lo;
        split_bf(fv[u], hi, lo);
        vThi[off] = hi;
        vTlo[off] = lo;
    }
}

// ---------------------------------------------------------------------------
// Flash attention, causal, bf16x3, one wave per (b,h, 16-row q tile).
// kv tiles of 32 (= WMMA K). Q/K split on the fly from f32 qkv; V from vT.
// Single wave per block -> prefetch next kv tile to hide global latency.
// ---------------------------------------------------------------------------
__global__ __launch_bounds__(32)
void attn_flash_bf16x3(const float* __restrict__ qkv,
                       const unsigned short* __restrict__ vThi,
                       const unsigned short* __restrict__ vTlo,
                       float* __restrict__ out, int S, int D, int H)
{
    const int row3 = 3 * D;
    const int lane = threadIdx.x;
    const int hf = lane >> 4, l16 = lane & 15;
    const int qt = blockIdx.x, h = blockIdx.y, b = blockIdx.z;
    const int bh = b * H + h;

    __shared__ __align__(16) unsigned short sPhi[16 * 32];
    __shared__ __align__(16) unsigned short sPlo[16 * 32];

    // Q fragments (A-layout), 2 K-chunks of 32 over hd=64.
    Frag qhi[2], qlo[2];
    {
        const float* qp = qkv + (size_t)(b * S + qt * 16 + l16) * row3 + h * 64;
        #pragma unroll
        for (int c = 0; c < 2; ++c) {
            #pragma unroll
            for (int g = 0; g < 2; ++g) {       // elems g*8..g*8+7 <- d = c*32+hf*16+g*8
                int d0 = c * 32 + hf * 16 + g * 8;
                float4 f0 = *reinterpret_cast<const float4*>(qp + d0);
                float4 f1 = *reinterpret_cast<const float4*>(qp + d0 + 4);
                float fv[8] = {f0.x, f0.y, f0.z, f0.w, f1.x, f1.y, f1.z, f1.w};
                #pragma unroll
                for (int u = 0; u < 8; ++u) {
                    unsigned short hi, lo;
                    split_bf(fv[u], hi, lo);
                    qhi[c].s[g * 8 + u] = hi;
                    qlo[c].s[g * 8 + u] = lo;
                }
            }
        }
    }

    v8f   o[4];
    float m[8], l[8];
    for (int j = 0; j < 4; ++j) o[j] = v8f{};
    for (int r = 0; r < 8; ++r) { m[r] = -3.0e38f; l[r] = 0.0f; }

    const int nkt = (qt >> 1) + 1;
    for (int kt = 0; kt < nkt; ++kt) {
        const int kv0 = kt * 32;

        // ---- prefetch next kv tile (K rows + vT slabs) ----
        if (kt + 1 < nkt) {
            __builtin_prefetch(
                qkv + (size_t)(b * S + kv0 + 32 + lane) * row3 + D + h * 64, 0, 1);
            __builtin_prefetch(vThi + ((size_t)bh * 64 + lane * 2) * S + kv0 + 32, 0, 1);
            __builtin_prefetch(vThi + ((size_t)bh * 64 + lane * 2 + 1) * S + kv0 + 32, 0, 1);
            __builtin_prefetch(vTlo + ((size_t)bh * 64 + lane * 2) * S + kv0 + 32, 0, 1);
            __builtin_prefetch(vTlo + ((size_t)bh * 64 + lane * 2 + 1) * S + kv0 + 32, 0, 1);
        }

        // ---- scores for 16x32 kv slab: two 16x16 accs, 12 WMMAs ----
        v8f s2[2]; s2[0] = v8f{}; s2[1] = v8f{};
        #pragma unroll
        for (int j = 0; j < 2; ++j) {
            const float* kp = qkv + (size_t)(b * S + kv0 + j * 16 + l16) * row3 + D + h * 64;
            #pragma unroll
            for (int c = 0; c < 2; ++c) {
                // B-frag: elems e=0..15 <- d = c*32 + hf*16 + e (contiguous)
                int d0 = c * 32 + hf * 16;
                float4 f0 = *reinterpret_cast<const float4*>(kp + d0);
                float4 f1 = *reinterpret_cast<const float4*>(kp + d0 + 4);
                float4 f2 = *reinterpret_cast<const float4*>(kp + d0 + 8);
                float4 f3 = *reinterpret_cast<const float4*>(kp + d0 + 12);
                float fv[16] = {f0.x,f0.y,f0.z,f0.w, f1.x,f1.y,f1.z,f1.w,
                                f2.x,f2.y,f2.z,f2.w, f3.x,f3.y,f3.z,f3.w};
                Frag kh, kl;
                #pragma unroll
                for (int u = 0; u < 16; ++u) {
                    unsigned short hi, lo;
                    split_bf(fv[u], hi, lo);
                    kh.s[u] = hi; kl.s[u] = lo;
                }
                s2[j] = wmma_bf16(qhi[c], kh, s2[j]);
                s2[j] = wmma_bf16(qhi[c], kl, s2[j]);
                s2[j] = wmma_bf16(qlo[c], kh, s2[j]);
            }
        }

        // ---- scale + causal mask (exact reference semantics) ----
        float sv0[8], sv1[8];
        #pragma unroll
        for (int r = 0; r < 8; ++r) {
            int row = qt * 16 + r + hf * 8;
            int c0 = kv0 + l16, c1 = kv0 + 16 + l16;
            sv0[r] = (c0 <= row) ? s2[0][r] * 0.125f : NEGV;
            sv1[r] = (c1 <= row) ? s2[1][r] * 0.125f : NEGV;
        }

        // ---- online softmax (butterflies stay inside 16-lane halves) ----
        #pragma unroll
        for (int r = 0; r < 8; ++r) {
            float tmax = fmaxf(sv0[r], sv1[r]);
            for (int off = 1; off < 16; off <<= 1)
                tmax = fmaxf(tmax, __shfl_xor(tmax, off, 32));
            float mnew  = fmaxf(m[r], tmax);
            float alpha = __expf(m[r] - mnew);
            float p0    = __expf(sv0[r] - mnew);
            float p1    = __expf(sv1[r] - mnew);
            float psum  = p0 + p1;
            for (int off = 1; off < 16; off <<= 1)
                psum += __shfl_xor(psum, off, 32);
            l[r] = l[r] * alpha + psum;
            m[r] = mnew;
            #pragma unroll
            for (int j = 0; j < 4; ++j) o[j][r] *= alpha;
            unsigned short hi, lo;
            split_bf(p0, hi, lo);
            sPhi[(r + hf * 8) * 32 + l16] = hi;
            sPlo[(r + hf * 8) * 32 + l16] = lo;
            split_bf(p1, hi, lo);
            sPhi[(r + hf * 8) * 32 + 16 + l16] = hi;
            sPlo[(r + hf * 8) * 32 + 16 + l16] = lo;
        }
        __syncthreads();

        // ---- O += P(16x32) * V(32x64): 12 WMMAs ----
        Frag pHi, pLo;
        pHi.q[0] = *reinterpret_cast<uint4*>(&sPhi[l16 * 32 + hf * 8]);
        pHi.q[1] = *reinterpret_cast<uint4*>(&sPhi[l16 * 32 + 16 + hf * 8]);
        pLo.q[0] = *reinterpret_cast<uint4*>(&sPlo[l16 * 32 + hf * 8]);
        pLo.q[1] = *reinterpret_cast<uint4*>(&sPlo[l16 * 32 + 16 + hf * 8]);
        #pragma unroll
        for (int j = 0; j < 4; ++j) {
            int n = j * 16 + l16;
            size_t voff = ((size_t)bh * 64 + n) * S + kv0 + hf * 8;
            Frag vh, vl;
            vh.q[0] = *reinterpret_cast<const uint4*>(vThi + voff);
            vh.q[1] = *reinterpret_cast<const uint4*>(vThi + voff + 16);
            vl.q[0] = *reinterpret_cast<const uint4*>(vTlo + voff);
            vl.q[1] = *reinterpret_cast<const uint4*>(vTlo + voff + 16);
            o[j] = wmma_bf16(pHi, vh, o[j]);
            o[j] = wmma_bf16(pHi, vl, o[j]);
            o[j] = wmma_bf16(pLo, vh, o[j]);
        }
        __syncthreads();
    }

    // ---- normalized output ----
    float* obase = out + (size_t)b * S * D + h * 64;
    for (int j = 0; j < 4; ++j) {
        int n = j * 16 + l16;
        for (int r = 0; r < 8; ++r) {
            int srow = qt * 16 + r + hf * 8;
            obase[(size_t)srow * D + n] = o[j][r] / l[r];
        }
    }
}

// string_len (int32) pass-through into tail of float d_out (bit copy).
__global__ void copy_len_kernel(const int* __restrict__ sl, float* __restrict__ out,
                                int off, int n)
{
    int i = threadIdx.x + blockIdx.x * blockDim.x;
    if (i < n) reinterpret_cast<int*>(out)[off + i] = sl[i];
}

extern "C" void kernel_launch(void* const* d_in, const int* in_sizes, int n_in,
                              void* d_out, int out_size, void* d_ws, size_t ws_size,
                              hipStream_t stream)
{
    const float* string     = (const float*)d_in[0];
    const int*   string_len = (const int*)  d_in[1];
    const float* W_attn     = (const float*)d_in[2];
    const float* b_attn     = (const float*)d_in[3];
    const float* W_proj     = (const float*)d_in[4];
    const float* b_proj     = (const float*)d_in[5];
    float*       out        = (float*)d_out;

    const int B = 4, S = 2048, D = 1024, H = 16;
    const int M = B * S;                                   // 8192

    char* ws = (char*)d_ws;
    float*          qkv   = (float*)ws;                    // 96 MB f32
    unsigned short* vThi  = (unsigned short*)(ws + (size_t)96  * 1024 * 1024);  // 16 MB
    unsigned short* vTlo  = (unsigned short*)(ws + (size_t)112 * 1024 * 1024);  // 16 MB
    float*          attn  = (float*)(ws + (size_t)128 * 1024 * 1024);           // 32 MB

    // 1) QKV projection (bf16x3)
    dim3 g1(M / 128, (3 * D) / 64);
    gemm_bf16x3<<<g1, 256, 0, stream>>>(string, W_attn, b_attn, qkv, M, 3 * D, D);

    // 2) V split + transpose for the P*V B-operand layout
    split_v_transpose<<<M, 256, 0, stream>>>(qkv, vThi, vTlo, S, D, H);

    // 3) causal flash attention (bf16x3)
    dim3 g2(S / 16, H, B);
    attn_flash_bf16x3<<<g2, 32, 0, stream>>>(qkv, vThi, vTlo, attn, S, D, H);

    // 4) output projection (bf16x3)
    dim3 g3(M / 128, D / 64);
    gemm_bf16x3<<<g3, 256, 0, stream>>>(attn, W_proj, b_proj, out, M, D, D);

    // 5) string_len tail
    copy_len_kernel<<<1, 32, 0, stream>>>(string_len, out, M * D, B);
}
